// MultiHeadSelfAttention_41102837023296
// MI455X (gfx1250) — matrix operand order repaired and checked by reading, not compiled, not run on hardware
//
#include <hip/hip_runtime.h>
#include <hip/hip_bf16.h>

typedef __attribute__((ext_vector_type(16))) _Float16 v16h;
typedef __attribute__((ext_vector_type(8)))  float    v8f;

#define NB    8
#define EMB   512
#define NHEAD 8
#define HDIM  64
#define SEQ   1024          // 32*32 spatial tokens
#define BROWS (NB*SEQ)      // 8192

// ---- CDNA5 async global->LDS copy (ASYNCcnt-tracked) -----------------------
__device__ __forceinline__ void async_g2l_b128(unsigned lds_byte_off,
                                               const void* gptr) {
  asm volatile("global_load_async_to_lds_b128 %0, %1, off"
               :: "v"(lds_byte_off),
                  "v"((unsigned long long)(size_t)gptr)
               : "memory");
}
__device__ __forceinline__ void wait_asynccnt0() {
#if __has_builtin(__builtin_amdgcn_s_wait_asynccnt)
  __builtin_amdgcn_s_wait_asynccnt(0);
#else
  asm volatile("s_wait_asynccnt 0x0" ::: "memory");
#endif
}

// ---- WMMA fragment loaders (per CDNA5 ISA 7.12.2 16-bit layouts) ------------
// A operand 16x32 (MxK): lane holds row M=lane%16; half=lane/16 selects K-phase.
__device__ __forceinline__ v16h frag_a(const _Float16* tile, int lane, int ld) {
  v16h a;
  const int m = lane & 15, half = lane >> 4;
  const _Float16* row = tile + m * ld;
#pragma unroll
  for (int v = 0; v < 8; ++v) {
    const int k0 = ((v & 4) << 2) + (half << 3) + ((v & 3) << 1);
    a[2*v]   = row[k0];
    a[2*v+1] = row[k0+1];
  }
  return a;
}

// B operand 32x16 (KxN) read from [ncol][k]-major storage (B transposed):
// lane holds col N=lane%16; VGPR v holds K = (v>=4)*16 + half*8 + 2(v&3), +1.
__device__ __forceinline__ v16h frag_b(const _Float16* bt, int lane, int ld) {
  v16h b;
  const int n = lane & 15, half = lane >> 4;
  const _Float16* col = bt + n * ld;
#pragma unroll
  for (int v = 0; v < 8; ++v) {
    const int k0 = ((v & 4) << 2) + (half << 3) + ((v & 3) << 1);
    b[2*v]   = col[k0];
    b[2*v+1] = col[k0+1];
  }
  return b;
}

// ---- QKV projection: dst[b,h,n,d] = f16( x_seq @ W^T + bias ) ---------------
// 256 threads = 8 waves, 128x64 tile. Transpose + f32->f16 fused into LDS load.
__global__ __launch_bounds__(256) void qkv_kernel(
    const float* __restrict__ x, const float* __restrict__ W,
    const float* __restrict__ bias, _Float16* __restrict__ dst) {
  __shared__ _Float16 At[128][34];
  __shared__ _Float16 Bt[64][34];
  const int t = threadIdx.x, lane = t & 31, w = t >> 5;
  const int half = lane >> 4, l16 = lane & 15;
  const int nBase = blockIdx.x * 64;          // output feature tile (one head)
  const int mBase = blockIdx.y * 128;         // token-row tile (within one batch)
  const int bIdx  = mBase >> 10;
  const int nLoc  = mBase & (SEQ - 1);
  const float* xb = x + (size_t)bIdx * EMB * SEQ;

  v8f acc[4];
#pragma unroll
  for (int tt = 0; tt < 4; ++tt) {
    const float bv = bias[nBase + tt*16 + l16];
#pragma unroll
    for (int r = 0; r < 8; ++r) acc[tt][r] = bv;
  }

  for (int k0 = 0; k0 < EMB; k0 += 32) {
    __syncthreads();
    // A tile: At[row][kk] = x[b][k0+kk][nLoc+row]  (coalesced over n)
#pragma unroll
    for (int i = 0; i < 16; ++i) {
      const int idx = t + i*256;
      const int row = idx & 127, kk = idx >> 7;
      At[row][kk] = (_Float16)xb[(size_t)(k0 + kk)*SEQ + nLoc + row];
    }
    // B tile: Bt[j][kk] = W[nBase+j][k0+kk]  (coalesced over k)
#pragma unroll
    for (int i = 0; i < 8; ++i) {
      const int idx = t + i*256;
      const int kk = idx & 31, j = idx >> 5;
      Bt[j][kk] = (_Float16)W[(size_t)(nBase + j)*EMB + k0 + kk];
    }
    // prefetch next k-slice while WMMAs run (global_prefetch_b8)
    if (k0 + 32 < EMB) {
      __builtin_prefetch(xb + (size_t)(k0 + 32 + (t >> 3))*SEQ + nLoc + (t & 7)*16, 0, 0);
      __builtin_prefetch(W + (size_t)(nBase + (t & 63))*EMB + k0 + 32, 0, 0);
    }
    __syncthreads();
    const v16h a = frag_a(&At[w*16][0], lane, 34);
    v16h bf[4];
#pragma unroll
    for (int tt = 0; tt < 4; ++tt) bf[tt] = frag_b(&Bt[tt*16][0], lane, 34);
#pragma unroll
    for (int tt = 0; tt < 4; ++tt)
      acc[tt] = __builtin_amdgcn_wmma_f32_16x16x32_f16(
          false, a, false, bf[tt], (short)0, acc[tt], false, false);
  }

  const int h = nBase >> 6;
#pragma unroll
  for (int tt = 0; tt < 4; ++tt) {
    const int dd = tt*16 + l16;
#pragma unroll
    for (int r = 0; r < 8; ++r) {
      const int gRow = mBase + w*16 + r + half*8;
      const int b = gRow >> 10, n = gRow & (SEQ-1);
      dst[(((size_t)(b*NHEAD + h)*SEQ) + n)*HDIM + dd] = (_Float16)acc[tt][r];
    }
  }
}

// ---- Flash attention with async double-buffered K tiles --------------------
__global__ __launch_bounds__(128) void attn_kernel(
    const _Float16* __restrict__ Q, const _Float16* __restrict__ Km,
    const _Float16* __restrict__ Vm, _Float16* __restrict__ Oc) {
  __shared__ _Float16 Kt[2][64][72];    // [buf][key][d]; 144B rows -> b128-aligned
  __shared__ _Float16 Vt[64][66];       // [d][key] (transposed, B-operand major)
  __shared__ _Float16 Ps[4][16][66];    // per-wave P staging for A-operand repack
  const int t = threadIdx.x, lane = t & 31, w = t >> 5;
  const int half = lane >> 4, l16 = lane & 15;
  const int bh = blockIdx.y;
  const int qBase = blockIdx.x * 64;
  const size_t base = (size_t)bh * SEQ * HDIM;
  const _Float16* Qp = Q  + base;
  const _Float16* Kp = Km + base;
  const _Float16* Vp = Vm + base;
  const int cj = t >> 1, cs = t & 1;    // copy mapping: row, 32-half segment

  // Q fragments for this wave's 16 rows (K-dim 64 = 2 frags of 32)
  v16h aq[2];
  {
    const _Float16* qrow = Qp + (size_t)(qBase + w*16 + l16) * HDIM;
#pragma unroll
    for (int f = 0; f < 2; ++f)
#pragma unroll
      for (int v = 0; v < 8; ++v) {
        const int k0 = f*32 + ((v & 4) << 2) + (half << 3) + ((v & 3) << 1);
        aq[f][2*v]   = qrow[k0];
        aq[f][2*v+1] = qrow[k0+1];
      }
  }

  float mrow[8], lrow[8];
  v8f O[4];
#pragma unroll
  for (int r = 0; r < 8; ++r) { mrow[r] = -3.0e38f; lrow[r] = 0.f; }
#pragma unroll
  for (int tt = 0; tt < 4; ++tt)
#pragma unroll
    for (int r = 0; r < 8; ++r) O[tt][r] = 0.f;

  // issue one K tile (64 keys x 64 d) as 4x b128 async copies per thread
  auto issueK = [&](int kb, int buf) {
    const _Float16* src = Kp + (size_t)(kb + cj)*HDIM + cs*32;
    const unsigned lds = (unsigned)(size_t)&Kt[buf][cj][cs*32];
#pragma unroll
    for (int o = 0; o < 4; ++o)
      async_g2l_b128(lds + (unsigned)(o*16), (const char*)src + o*16);
  };
  issueK(0, 0);

  for (int it = 0; it < SEQ/64; ++it) {
    const int kb = it * 64;
    const int cur = it & 1;
    // stage V tile to registers early (overlaps with async K + wait)
    unsigned vreg[16];
    {
      const unsigned* vrow =
          (const unsigned*)(Vp + (size_t)(kb + cj)*HDIM + cs*32);
#pragma unroll
      for (int q = 0; q < 16; ++q) vreg[q] = vrow[q];
    }
    wait_asynccnt0();            // my Kt[cur] copies landed
    __syncthreads();             // everyone's landed; all waves done with it-1
    if (it + 1 < SEQ/64) issueK(kb + 64, cur ^ 1);   // overlap next K tile
    // V transpose -> LDS
#pragma unroll
    for (int q = 0; q < 16; ++q) {
      Vt[cs*32 + 2*q + 0][cj] = ((const _Float16*)&vreg[q])[0];
      Vt[cs*32 + 2*q + 1][cj] = ((const _Float16*)&vreg[q])[1];
    }
    __syncthreads();

    // S = (Q K^T) * 1/sqrt(64)
    v8f S[4];
#pragma unroll
    for (int tt = 0; tt < 4; ++tt) {
      v8f s;
#pragma unroll
      for (int r = 0; r < 8; ++r) s[r] = 0.f;
      const v16h b0 = frag_b(&Kt[cur][tt*16][0], lane, 72);
      s = __builtin_amdgcn_wmma_f32_16x16x32_f16(false, aq[0], false, b0, (short)0, s, false, false);
      const v16h b1 = frag_b(&Kt[cur][tt*16][0] + 32, lane, 72);
      s = __builtin_amdgcn_wmma_f32_16x16x32_f16(false, aq[1], false, b1, (short)0, s, false, false);
#pragma unroll
      for (int r = 0; r < 8; ++r) s[r] *= 0.125f;
      S[tt] = s;
    }

    // Online softmax: row M's 16 cols live in one 16-lane half-group.
#pragma unroll
    for (int r = 0; r < 8; ++r) {
      float v = fmaxf(fmaxf(S[0][r], S[1][r]), fmaxf(S[2][r], S[3][r]));
#pragma unroll
      for (int off = 1; off < 16; off <<= 1)
        v = fmaxf(v, __shfl_xor(v, off, 16));
      const float mn    = fmaxf(mrow[r], v);
      const float alpha = __expf(mrow[r] - mn);
      mrow[r] = mn;
      lrow[r] *= alpha;
#pragma unroll
      for (int tt = 0; tt < 4; ++tt) O[tt][r] *= alpha;
      float rs = 0.f;
#pragma unroll
      for (int tt = 0; tt < 4; ++tt) {
        const float p = __expf(S[tt][r] - mn);
        rs += p;
        Ps[w][r + half*8][tt*16 + l16] = (_Float16)p;  // C/D layout -> LDS
      }
#pragma unroll
      for (int off = 1; off < 16; off <<= 1) rs += __shfl_xor(rs, off, 16);
      lrow[r] += rs;
    }

    // O += P V  (same-wave DS ops are in-order; Ps is wave-private)
    const v16h ap0 = frag_a(&Ps[w][0][0], lane, 66);
    const v16h ap1 = frag_a(&Ps[w][0][0] + 32, lane, 66);
#pragma unroll
    for (int tt = 0; tt < 4; ++tt) {
      const v16h b0 = frag_b(&Vt[tt*16][0], lane, 66);
      O[tt] = __builtin_amdgcn_wmma_f32_16x16x32_f16(false, ap0, false, b0, (short)0, O[tt], false, false);
      const v16h b1 = frag_b(&Vt[tt*16][0] + 32, lane, 66);
      O[tt] = __builtin_amdgcn_wmma_f32_16x16x32_f16(false, ap1, false, b1, (short)0, O[tt], false, false);
    }
  }

  // Normalize and store heads-concatenated: Oc[b][n][h*64+dd]
  const int b = bh >> 3, h = bh & 7;
#pragma unroll
  for (int tt = 0; tt < 4; ++tt) {
#pragma unroll
    for (int r = 0; r < 8; ++r) {
      const int n = qBase + w*16 + r + half*8;
      const int c = h*HDIM + tt*16 + l16;
      Oc[((size_t)b*SEQ + n)*EMB + c] = (_Float16)(O[tt][r] / lrow[r]);
    }
  }
}

// ---- Output projection + residual: y[b,c,n] = gamma*(Oc Wo^T + bo) + x -----
__global__ __launch_bounds__(256) void out_kernel(
    const _Float16* __restrict__ Oc, const float* __restrict__ Wo,
    const float* __restrict__ bo, const float* __restrict__ x,
    const float* __restrict__ gamma, float* __restrict__ y) {
  __shared__ _Float16 At[128][34];
  __shared__ _Float16 Bt[64][34];
  const int t = threadIdx.x, lane = t & 31, w = t >> 5;
  const int half = lane >> 4, l16 = lane & 15;
  const int nBase = blockIdx.x * 64;   // output channel tile
  const int mBase = blockIdx.y * 128;  // token rows

  v8f acc[4];
#pragma unroll
  for (int tt = 0; tt < 4; ++tt) {
    const float bv = bo[nBase + tt*16 + l16];
#pragma unroll
    for (int r = 0; r < 8; ++r) acc[tt][r] = bv;
  }

  for (int k0 = 0; k0 < EMB; k0 += 32) {
    __syncthreads();
#pragma unroll
    for (int i = 0; i < 16; ++i) {
      const int idx = t + i*256;
      const int kk = idx & 31, row = idx >> 5;
      At[row][kk] = Oc[(size_t)(mBase + row)*EMB + k0 + kk];
    }
#pragma unroll
    for (int i = 0; i < 8; ++i) {
      const int idx = t + i*256;
      const int kk = idx & 31, j = idx >> 5;
      Bt[j][kk] = (_Float16)Wo[(size_t)(nBase + j)*EMB + k0 + kk];
    }
    if (k0 + 32 < EMB) {
      __builtin_prefetch(Oc + (size_t)(mBase + (t >> 1))*EMB + k0 + 32 + (t & 1)*16, 0, 0);
      __builtin_prefetch(Wo + (size_t)(nBase + (t & 63))*EMB + k0 + 32, 0, 0);
    }
    __syncthreads();
    const v16h a = frag_a(&At[w*16][0], lane, 34);
    v16h bf[4];
#pragma unroll
    for (int tt = 0; tt < 4; ++tt) bf[tt] = frag_b(&Bt[tt*16][0], lane, 34);
#pragma unroll
    for (int tt = 0; tt < 4; ++tt)
      acc[tt] = __builtin_amdgcn_wmma_f32_16x16x32_f16(
          false, a, false, bf[tt], (short)0, acc[tt], false, false);
  }

  const float g = gamma[0];
#pragma unroll
  for (int tt = 0; tt < 4; ++tt) {
    const int c = nBase + tt*16 + l16;
#pragma unroll
    for (int r = 0; r < 8; ++r) {
      const int gRow = mBase + w*16 + r + half*8;
      const int b = gRow >> 10, n = gRow & (SEQ-1);
      const size_t xi = ((size_t)b*EMB + c)*SEQ + n;
      y[xi] = g*acc[tt][r] + x[xi];
    }
  }
}

extern "C" void kernel_launch(void* const* d_in, const int* in_sizes, int n_in,
                              void* d_out, int out_size, void* d_ws, size_t ws_size,
                              hipStream_t stream) {
  (void)in_sizes; (void)n_in; (void)out_size; (void)ws_size;
  const float* x     = (const float*)d_in[0];
  const float* Wq    = (const float*)d_in[1];
  const float* bq    = (const float*)d_in[2];
  const float* Wk    = (const float*)d_in[3];
  const float* bk    = (const float*)d_in[4];
  const float* Wv    = (const float*)d_in[5];
  const float* bv    = (const float*)d_in[6];
  const float* Wo    = (const float*)d_in[7];
  const float* bo    = (const float*)d_in[8];
  const float* gamma = (const float*)d_in[9];
  float* y = (float*)d_out;

  // workspace: Q,K,V [b,h,n,d] f16 (8 MB each) + Oc [b,n,c] f16 (8 MB) = 32 MB
  const size_t qkvElems = (size_t)NB * NHEAD * SEQ * HDIM;
  _Float16* Q  = (_Float16*)d_ws;
  _Float16* K  = Q + qkvElems;
  _Float16* V  = K + qkvElems;
  _Float16* Oc = V + qkvElems;

  dim3 gGemm(EMB/64, BROWS/128);                // (8, 64)
  qkv_kernel<<<gGemm, dim3(256), 0, stream>>>(x, Wq, bq, Q);
  qkv_kernel<<<gGemm, dim3(256), 0, stream>>>(x, Wk, bk, K);
  qkv_kernel<<<gGemm, dim3(256), 0, stream>>>(x, Wv, bv, V);
  attn_kernel<<<dim3(SEQ/64, NB*NHEAD), dim3(128), 0, stream>>>(Q, K, V, Oc);
  out_kernel<<<gGemm, dim3(256), 0, stream>>>(Oc, Wo, bo, x, gamma, y);
}
